// GATv2ConvWithEdgeFeat_21741124452820
// MI455X (gfx1250) — matrix operand order repaired and checked by reading, not compiled, not run on hardware
//
#include <hip/hip_runtime.h>
#include <hip/hip_bf16.h>

// ---------------------------------------------------------------------------
// GATv2 with edge features, MI455X (gfx1250, wave32, WMMA).
//
// Pipeline (all on `stream`):
//   0) init:       zero d_out, zero z, init softmax-max keys
//   1) proj x2:    feat_src = x@W_src^T+b_src, feat_dst = x@W_dst^T+b_dst  (WMMA bf16)
//   2) edge_score: persistent blocks; per 64-edge tile: feat_edge via WMMA -> LDS,
//                  score = LeakyReLU(fs+fd+fe).attn, atomicMax per (dst,head)
//   3) edge_norm:  ex = exp(score - m[dst]); z[dst] += ex
//   4) edge_msg:   persistent; recompute feat_edge tile, out[dst] += (fs+fe)*ex/z
//   5) relu in place on d_out
//
// Design points for MI455X:
//  * feat_edge never materialized in HBM (saves 410MB ws + 1.2GB traffic);
//    efeat streamed twice (2 x 410MB ~= 35us at 23.3 TB/s).
//  * bf16 WMMA (f32 accumulate) keeps the 26-GFLOP edge GEMM far under the
//    matrix-core roofline, so the kernel stays HBM-bound.
//  * Persistent edge blocks: W_edge staged to LDS ONCE per block (vs per tile),
//    and A tiles are double-buffered: next tile's efeat rows are fetched into
//    VGPRs (global b128) while WMMAs run, then cvt->bf16->LDS. Plain VMEM
//    staging is intentional: async-to-LDS/TDM can't do the f32->bf16 convert.
//  * __builtin_prefetch (-> global_prefetch_b8) warms next tile's src/dst.
// ---------------------------------------------------------------------------

typedef __attribute__((ext_vector_type(16))) __bf16 v16bf;
typedef __attribute__((ext_vector_type(8)))  float  v8f;

union Frag {
    v16bf    v;
    unsigned u[8];   // 8 dwords = 16 packed bf16
};

#define LDS_KPAIR 68   // 64 K-pairs (128 bf16) + 4 pad dwords -> bank-conflict-free
#define FE_STRIDE 132  // 128 f32 + 4 pad
#define PERSIST_BLOCKS 2048

// float -> bf16 (round to nearest even), packed as u16
__device__ __forceinline__ unsigned f2bf(float f) {
    unsigned u = __float_as_uint(f);
    return (u + 0x7FFFu + ((u >> 16) & 1u)) >> 16;
}

// order-preserving float<->uint map for atomicMax-based segment max
__device__ __forceinline__ unsigned fkey(float f) {
    unsigned u = __float_as_uint(f);
    return (u & 0x80000000u) ? ~u : (u | 0x80000000u);
}
__device__ __forceinline__ float key2f(unsigned k) {
    unsigned u = (k & 0x80000000u) ? (k & 0x7FFFFFFFu) : ~k;
    return __uint_as_float(u);
}

// Cooperative load of nRows x 128 f32 matrix -> LDS as packed bf16 pairs.
__device__ __forceinline__ void load_matrix_bf16(const float* __restrict__ G,
                                                 int rowBase, int nRows, int M,
                                                 unsigned* sDst) {
    for (int i = threadIdx.x; i < nRows * 32; i += blockDim.x) {
        int r   = i >> 5;
        int c4  = i & 31;
        int row = rowBase + r;
        float4 v = make_float4(0.f, 0.f, 0.f, 0.f);
        if (row < M) v = ((const float4*)G)[(size_t)row * 32 + c4];
        sDst[r * LDS_KPAIR + c4 * 2]     = f2bf(v.x) | (f2bf(v.y) << 16);
        sDst[r * LDS_KPAIR + c4 * 2 + 1] = f2bf(v.z) | (f2bf(v.w) << 16);
    }
}

// Fetch one 64x128 f32 tile into per-thread VGPRs (8 x float4, coalesced b128).
__device__ __forceinline__ void tile_fetch(const float* __restrict__ G,
                                           int rowBase, int M, float4 v[8]) {
#pragma unroll
    for (int j = 0; j < 8; ++j) {
        int idx = threadIdx.x + 256 * j;
        int r = idx >> 5, c4 = idx & 31;
        int row = rowBase + r;
        v[j] = make_float4(0.f, 0.f, 0.f, 0.f);
        if (row < M) v[j] = ((const float4*)G)[(size_t)row * 32 + c4];
    }
}

// Convert staged VGPR tile -> packed-bf16 LDS buffer.
__device__ __forceinline__ void tile_commit(const float4 v[8], unsigned* sDst) {
#pragma unroll
    for (int j = 0; j < 8; ++j) {
        int idx = threadIdx.x + 256 * j;
        int r = idx >> 5, c4 = idx & 31;
        sDst[r * LDS_KPAIR + c4 * 2]     = f2bf(v[j].x) | (f2bf(v[j].y) << 16);
        sDst[r * LDS_KPAIR + c4 * 2 + 1] = f2bf(v[j].z) | (f2bf(v[j].w) << 16);
    }
}

// One 64x128 output tile: acc[mt] = A[64x128] x W^T column block (wave w owns
// cols [16w,16w+16)). K=128 -> 4 WMMA steps. A and B in contiguous k order (k = 16h + s).
__device__ __forceinline__ void mma64x128(const unsigned* sA, const unsigned* sW,
                                          v8f acc[4]) {
    const int lane = threadIdx.x & 31;
    const int wave = (threadIdx.x >> 5) & 7;

    Frag bf[4];
    {
        const int o   = wave * 16 + (lane & 15);
        const int bko = (lane >> 4) * 8;
#pragma unroll
        for (int kb = 0; kb < 4; ++kb) {
            int base = o * LDS_KPAIR + kb * 16 + bko;
#pragma unroll
            for (int j = 0; j < 8; ++j) bf[kb].u[j] = sW[base + j];
        }
    }

#pragma unroll
    for (int mt = 0; mt < 4; ++mt) {
        v8f c = {};
        const int m   = mt * 16 + (lane & 15);
        const int ka0 = (lane >> 4) * 8;
#pragma unroll
        for (int kb = 0; kb < 4; ++kb) {
            Frag af;
            int base = m * LDS_KPAIR + kb * 16 + ka0;
#pragma unroll
            for (int j = 0; j < 8; ++j) {
                af.u[j]     = sA[base + j];
            }
            c = __builtin_amdgcn_wmma_f32_16x16x32_bf16(
                    false, af.v, false, bf[kb].v, (short)0, c, false, false);
        }
        acc[mt] = c;
    }
}

__device__ __forceinline__ void acc_to_lds(const v8f acc[4], float* sFE) {
    const int lane = threadIdx.x & 31;
    const int wave = threadIdx.x >> 5;
    const int col  = wave * 16 + (lane & 15);
#pragma unroll
    for (int mt = 0; mt < 4; ++mt) {
        int r0 = mt * 16 + (lane >> 4) * 8;
#pragma unroll
        for (int r = 0; r < 8; ++r) sFE[(r0 + r) * FE_STRIDE + col] = acc[mt][r];
    }
}

// ---------------------------------------------------------------------------

__global__ void init_kernel(float* __restrict__ out, unsigned* __restrict__ mkey,
                            float* __restrict__ z, int N) {
    int i = blockIdx.x * blockDim.x + threadIdx.x;
    if (i < N * 128) out[i] = 0.f;
    if (i < N * 8) { mkey[i] = 0u; z[i] = 0.f; }  // key 0 < key of any finite float
}

// C[M,128] = A[M,128] @ W[128,128]^T + bias
__global__ __launch_bounds__(256) void proj_kernel(
        const float* __restrict__ A, const float* __restrict__ Wm,
        const float* __restrict__ bias, float* __restrict__ C, int M) {
    __shared__ unsigned sW[128 * LDS_KPAIR];
    __shared__ unsigned sA[64 * LDS_KPAIR];
    const int rowBase = blockIdx.x * 64;

    load_matrix_bf16(Wm, 0, 128, 128, sW);
    load_matrix_bf16(A, rowBase, 64, M, sA);
    __syncthreads();

    v8f acc[4];
    mma64x128(sA, sW, acc);

    const int lane = threadIdx.x & 31;
    const int wave = threadIdx.x >> 5;
    const int col  = wave * 16 + (lane & 15);
    const float b  = bias[col];
#pragma unroll
    for (int mt = 0; mt < 4; ++mt) {
        int row0 = rowBase + mt * 16 + (lane >> 4) * 8;
#pragma unroll
        for (int r = 0; r < 8; ++r) {
            int row = row0 + r;
            if (row < M) C[(size_t)row * 128 + col] = acc[mt][r] + b;
        }
    }
}

// persistent + double-buffered: feat_edge tile (WMMA->LDS), scores, segment max
__global__ __launch_bounds__(256) void edge_score_kernel(
        const float* __restrict__ EF, const float* __restrict__ Wm,
        const float* __restrict__ attn,
        const int* __restrict__ src, const int* __restrict__ dst,
        const float* __restrict__ featS, const float* __restrict__ featD,
        float* __restrict__ scoreWS, unsigned* __restrict__ mkey,
        int E, int nTiles) {
    __shared__ unsigned sW[128 * LDS_KPAIR];
    __shared__ unsigned sA[2][64 * LDS_KPAIR];
    __shared__ float    sFE[64 * FE_STRIDE];
    __shared__ float    sAttn[128];

    if ((int)blockIdx.x >= nTiles) return;
    load_matrix_bf16(Wm, 0, 128, 128, sW);               // once per block
    if (threadIdx.x < 128) sAttn[threadIdx.x] = attn[threadIdx.x];

    const int stride = gridDim.x;
    float4 stg[8];
    tile_fetch(EF, blockIdx.x * 64, E, stg);
    int cur = 0;

    for (int t = blockIdx.x; t < nTiles; t += stride) {
        tile_commit(stg, sA[cur]);
        const int tn = t + stride;
        if (tn < nTiles) {
            __builtin_prefetch(src + tn * 64, 0, 3);     // -> global_prefetch_b8
            __builtin_prefetch(dst + tn * 64, 0, 3);
            tile_fetch(EF, tn * 64, E, stg);             // overlap with WMMAs below
        }
        __syncthreads();

        v8f acc[4];
        mma64x128(sA[cur], sW, acc);
        acc_to_lds(acc, sFE);
        __syncthreads();

        const int e0 = t * 64;
        for (int p = threadIdx.x; p < 512; p += 256) {   // 64 edges x 8 heads
            int el = p >> 3, h = p & 7;
            int e = e0 + el;
            if (e < E) {
                int s = src[e], d = dst[e];
                const float4* fsp = (const float4*)(featS + (size_t)s * 128 + h * 16);
                const float4* fdp = (const float4*)(featD + (size_t)d * 128 + h * 16);
                const float*  fe  = &sFE[el * FE_STRIDE + h * 16];
                float sc = 0.f;
#pragma unroll
                for (int i = 0; i < 4; ++i) {
                    float4 a4 = fsp[i], b4 = fdp[i];
                    float tv;
                    tv = a4.x + b4.x + fe[4 * i + 0]; tv = (tv >= 0.f) ? tv : 0.2f * tv; sc += tv * sAttn[h * 16 + 4 * i + 0];
                    tv = a4.y + b4.y + fe[4 * i + 1]; tv = (tv >= 0.f) ? tv : 0.2f * tv; sc += tv * sAttn[h * 16 + 4 * i + 1];
                    tv = a4.z + b4.z + fe[4 * i + 2]; tv = (tv >= 0.f) ? tv : 0.2f * tv; sc += tv * sAttn[h * 16 + 4 * i + 2];
                    tv = a4.w + b4.w + fe[4 * i + 3]; tv = (tv >= 0.f) ? tv : 0.2f * tv; sc += tv * sAttn[h * 16 + 4 * i + 3];
                }
                scoreWS[(size_t)e * 8 + h] = sc;
                atomicMax(&mkey[(size_t)d * 8 + h], fkey(sc));
            }
        }
        __syncthreads();
        cur ^= 1;
    }
}

__global__ void edge_norm_kernel(float* __restrict__ scoreWS,
                                 const int* __restrict__ dst,
                                 const unsigned* __restrict__ mkey,
                                 float* __restrict__ z, int E) {
    int i = blockIdx.x * blockDim.x + threadIdx.x;
    if (i >= E * 8) return;
    int e = i >> 3, h = i & 7;
    int d = dst[e];
    float m  = key2f(mkey[(size_t)d * 8 + h]);
    float ex = __expf(scoreWS[i] - m);
    scoreWS[i] = ex;
    atomicAdd(&z[(size_t)d * 8 + h], ex);
}

// persistent + double-buffered: feat_edge tile, weighted scatter of (fs+fe)*a
__global__ __launch_bounds__(256) void edge_msg_kernel(
        const float* __restrict__ EF, const float* __restrict__ Wm,
        const int* __restrict__ src, const int* __restrict__ dst,
        const float* __restrict__ featS, const float* __restrict__ scoreWS,
        const float* __restrict__ z, float* __restrict__ out,
        int E, int nTiles) {
    __shared__ unsigned sW[128 * LDS_KPAIR];
    __shared__ unsigned sA[2][64 * LDS_KPAIR];
    __shared__ float    sFE[64 * FE_STRIDE];

    if ((int)blockIdx.x >= nTiles) return;
    load_matrix_bf16(Wm, 0, 128, 128, sW);               // once per block

    const int stride = gridDim.x;
    float4 stg[8];
    tile_fetch(EF, blockIdx.x * 64, E, stg);
    int cur = 0;

    for (int t = blockIdx.x; t < nTiles; t += stride) {
        tile_commit(stg, sA[cur]);
        const int tn = t + stride;
        if (tn < nTiles) {
            __builtin_prefetch(src + tn * 64, 0, 3);
            __builtin_prefetch(dst + tn * 64, 0, 3);
            tile_fetch(EF, tn * 64, E, stg);
        }
        __syncthreads();

        v8f acc[4];
        mma64x128(sA[cur], sW, acc);
        acc_to_lds(acc, sFE);
        __syncthreads();

        const int e0 = t * 64;
        for (int p = threadIdx.x; p < 512; p += 256) {
            int el = p >> 3, h = p & 7;
            int e = e0 + el;
            if (e < E) {
                int s = src[e], d = dst[e];
                float a = scoreWS[(size_t)e * 8 + h] / z[(size_t)d * 8 + h];
                const float4* fsp = (const float4*)(featS + (size_t)s * 128 + h * 16);
                const float*  fe  = &sFE[el * FE_STRIDE + h * 16];
                float* op = out + (size_t)d * 128 + h * 16;
#pragma unroll
                for (int i = 0; i < 4; ++i) {
                    float4 a4 = fsp[i];
                    atomicAdd(&op[4 * i + 0], (a4.x + fe[4 * i + 0]) * a);
                    atomicAdd(&op[4 * i + 1], (a4.y + fe[4 * i + 1]) * a);
                    atomicAdd(&op[4 * i + 2], (a4.z + fe[4 * i + 2]) * a);
                    atomicAdd(&op[4 * i + 3], (a4.w + fe[4 * i + 3]) * a);
                }
            }
        }
        __syncthreads();
        cur ^= 1;
    }
}

__global__ void relu_kernel(float* __restrict__ out, int N) {
    int i = blockIdx.x * blockDim.x + threadIdx.x;
    if (i < N * 128) out[i] = fmaxf(out[i], 0.f);
}

// ---------------------------------------------------------------------------

extern "C" void kernel_launch(void* const* d_in, const int* in_sizes, int n_in,
                              void* d_out, int out_size, void* d_ws, size_t ws_size,
                              hipStream_t stream) {
    const float* x      = (const float*)d_in[0];
    const float* efeat  = (const float*)d_in[1];
    const int*   src    = (const int*)d_in[2];
    const int*   dst    = (const int*)d_in[3];
    const float* W_src  = (const float*)d_in[4];
    const float* b_src  = (const float*)d_in[5];
    const float* W_dst  = (const float*)d_in[6];
    const float* b_dst  = (const float*)d_in[7];
    const float* W_edge = (const float*)d_in[8];
    const float* attn   = (const float*)d_in[9];
    float*       out    = (float*)d_out;

    const int N = in_sizes[0] / 128;   // 50000
    const int E = in_sizes[2];         // 800000

    // workspace carve-out (~80 MB): featS | featD | score/ex | mkey | z
    float*    featS   = (float*)d_ws;
    float*    featD   = featS + (size_t)N * 128;
    float*    scoreWS = featD + (size_t)N * 128;
    unsigned* mkey    = (unsigned*)(scoreWS + (size_t)E * 8);
    float*    z       = (float*)(mkey + (size_t)N * 8);

    const dim3 blk(256);
    const int nodeBlocks = (N + 63) / 64;
    const int nTiles     = (E + 63) / 64;
    const int edgeBlocks = (nTiles < PERSIST_BLOCKS) ? nTiles : PERSIST_BLOCKS;

    init_kernel<<<(N * 128 + 255) / 256, blk, 0, stream>>>(out, mkey, z, N);
    proj_kernel<<<nodeBlocks, blk, 0, stream>>>(x, W_src, b_src, featS, N);
    proj_kernel<<<nodeBlocks, blk, 0, stream>>>(x, W_dst, b_dst, featD, N);
    edge_score_kernel<<<edgeBlocks, blk, 0, stream>>>(efeat, W_edge, attn, src, dst,
                                                      featS, featD, scoreWS, mkey,
                                                      E, nTiles);
    edge_norm_kernel<<<(E * 8 + 255) / 256, blk, 0, stream>>>(scoreWS, dst, mkey, z, E);
    edge_msg_kernel<<<edgeBlocks, blk, 0, stream>>>(efeat, W_edge, src, dst,
                                                    featS, scoreWS, z, out,
                                                    E, nTiles);
    relu_kernel<<<(N * 128 + 255) / 256, blk, 0, stream>>>(out, N);
}
